// BandSplit_91173565760186
// MI455X (gfx1250) — matrix operand order repaired and checked by reading, not compiled, not run on hardware
//
#include <hip/hip_runtime.h>

// ---------------------------------------------------------------------------
// BandSplit for MI455X (gfx1250): 64 per-band GEMMs via v_wmma_f32_16x16x32_f16
//   FLOPs ~34G, min traffic ~0.4GB.
//   Pass 1: fold mel weights into pre_w, cast f16                      (tiny)
//   Pass 2: per (band, 256-row M tile) GEMM -> scratch (b,k,t,o) coalesced
//           * wave = M32 x N128: 16 wmma per (8 B + 2 A) LDS frag loads,
//             halving B-fragment LDS traffic vs M16 (LDS-bandwidth balanced)
//           * gather offsets precomputed branchlessly in LDS; staging issues
//             32 independent global loads per thread (full MLP, unrolled)
//   Pass 3: LDS-tiled (k,o)->(o,k) transpose into the botk output
// ---------------------------------------------------------------------------

typedef _Float16 v16h __attribute__((ext_vector_type(16)));
typedef _Float16 h8   __attribute__((ext_vector_type(8)));
typedef float    v8f  __attribute__((ext_vector_type(8)));

union F16Frag { v16h v; h8 h[2]; };

#define T_DIM    1024
#define FBINS    1025
#define INCH     2
#define OUTCH    128
#define NBANDS   64
#define MTILE    256     // rows per workgroup (8 waves x 32 rows)
#define AROW     40      // A tile row stride in halves (32 + 8 pad -> conflict-free)
#define BROW     136     // B chunk row stride in halves (128 + 8 pad)
#define KCHUNK   128     // K extent of the B panel staged in LDS at a time
#define KPAD_MAX 2080    // W <= FBINS => Kdim = 2W <= 2050, padded to 32

// ---- Pass 1: Wc[band][o][kk] = f16( mel_w[band][w] * pre_w[band][c][w][o] ),
//      kk = c*W + w, zero padded to Kpad. Zero weights neutralize padded taps.
__global__ __launch_bounds__(256) void prep_weights(
    const float* __restrict__ mel_w, const float* __restrict__ pre_w,
    _Float16* __restrict__ wc, int W, int Kpad, int total) {
  int e = blockIdx.x * 256 + threadIdx.x;
  if (e >= total) return;
  int kk   = e % Kpad;
  int rem  = e / Kpad;
  int o    = rem & (OUTCH - 1);
  int band = rem >> 7;
  float v = 0.0f;
  int Kdim = 2 * W;
  if (kk < Kdim) {
    int c = (kk >= W) ? 1 : 0;
    int w = kk - (c ? W : 0);
    v = mel_w[band * W + w] *
        pre_w[((size_t)((band * INCH + c) * W + w)) * OUTCH + o];
  }
  wc[e] = (_Float16)v;
}

// ---- Pass 2: per-band GEMM. Block = 256 threads = 8 wave32.
// Wave w owns M rows [32w,32w+32) (two A frags), all N=128 (8 B frags).
__global__ __launch_bounds__(256) void band_gemm(
    const float* __restrict__ x, const int* __restrict__ idx,
    const _Float16* __restrict__ wc, const float* __restrict__ pre_b,
    float* __restrict__ scratch, int W, int Kpad) {
  __shared__ _Float16 smem[MTILE * AROW + OUTCH * BROW];
  __shared__ int offLds[KPAD_MAX];       // gather offsets into x (built once)
  _Float16* aLds = smem;                 // 256 x 32 f16 A tile (gathered)
  _Float16* bLds = smem + MTILE * AROW;  // 128 x KCHUNK f16 B panel chunk

  const int mtile = blockIdx.x;   // 0..15 (M = B*T = 4096, tiles of 256)
  const int band  = blockIdx.y;   // 0..63
  const int tid   = threadIdx.x;
  const int wave  = tid >> 5;
  const int lane  = tid & 31;
  const int n     = lane & 15;
  const int hi    = lane >> 4;
  const int Kdim  = 2 * W;

  // ---- branchless gather-offset table (idx read from global ONCE per block).
  // Padded taps point at element 0: in-bounds, and their folded weight is 0.
  {
    const int* bandIdx = idx + band * W;
    for (int kg = tid; kg < Kpad; kg += 256) {
      int off = 0;
      if (kg < Kdim) {
        int c = (kg >= W) ? 1 : 0;
        int w = kg - (c ? W : 0);
        off = bandIdx[w] + (c ? T_DIM * FBINS : 0);  // channel folded in
      }
      offLds[kg] = off;
    }
  }

  // accumulators, initialized with bias (same bias for every row of a frag)
  v8f acc0[8], acc1[8];
#pragma unroll
  for (int nt = 0; nt < 8; ++nt) {
    float bv = pre_b[band * OUTCH + nt * 16 + n];
    v8f c;
#pragma unroll
    for (int i = 0; i < 8; ++i) c[i] = bv;
    acc0[nt] = c;
    acc1[nt] = c;
  }

  // A staging: one thread per row, all 32 k's of the step
  const int growA = mtile * MTILE + tid;
  const float* xrow =
      x + ((size_t)((growA >> 10) * INCH) * T_DIM + (growA & (T_DIM - 1))) * FBINS;
  _Float16* aRow = aLds + tid * AROW;

  // B staging: thread -> fixed row, strided 16B chunks (no division)
  const int rB  = tid & 127;
  const int cv0 = tid >> 7;  // 0 or 1
  const _Float16* wrow = wc + (size_t)(band * OUTCH + rB) * Kpad;

  for (int kc = 0; kc < Kpad; kc += KCHUNK) {
    const int clen = (Kpad - kc < KCHUNK) ? (Kpad - kc) : KCHUNK;  // mult of 32
    const int vrow = clen >> 3;
    for (int cv = cv0; cv < vrow; cv += 2)
      *(h8*)(bLds + rB * BROW + cv * 8) = *(const h8*)(wrow + kc + cv * 8);
    __syncthreads();  // first iteration: also covers offLds

    for (int ks = 0; ks < clen; ks += 32) {
      const int kbase = kc + ks;
      // ---- gather A tile 256x32: vector-load 32 offsets, issue 32
      // independent global loads (full MLP, single wait domain), pack 4x b128.
      int offs[32];
#pragma unroll
      for (int g = 0; g < 8; ++g) {
        int4 o4 = *(const int4*)(offLds + kbase + g * 4);
        offs[g*4+0] = o4.x; offs[g*4+1] = o4.y;
        offs[g*4+2] = o4.z; offs[g*4+3] = o4.w;
      }
      float vals[32];
#pragma unroll
      for (int i = 0; i < 32; ++i) vals[i] = xrow[offs[i]];
#pragma unroll
      for (int g = 0; g < 4; ++g) {
        h8 p;
#pragma unroll
        for (int i = 0; i < 8; ++i) p[i] = (_Float16)vals[g * 8 + i];
        *(h8*)(aRow + g * 8) = p;
      }
      __syncthreads();

      // ---- A fragments (ISA 16-bit A 16x32 layout):
      //   lanes 0-15 : M=lane,    K = 0..7  and 16..23
      //   lanes 16-31: M=lane-16, K = 8..15 and 24..31
      F16Frag a0, a1;
      {
        const _Float16* ap = aLds + (wave * 32 + n) * AROW + (hi ? 8 : 0);
        a0.h[0] = *(const h8*)ap;
        a0.h[1] = *(const h8*)(ap + 16);
        a1.h[0] = *(const h8*)(ap + 16 * AROW);
        a1.h[1] = *(const h8*)(ap + 16 * AROW + 16);
      }
      // ---- 8 N-tiles: one B fragment feeds two WMMAs (halves B LDS traffic)
#pragma unroll
      for (int nt = 0; nt < 8; ++nt) {
        F16Frag bf;
        const _Float16* bp = bLds + (nt * 16 + n) * BROW + ks + (hi ? 8 : 0);
        bf.h[0] = *(const h8*)bp;
        bf.h[1] = *(const h8*)(bp + 16);
        acc0[nt] = __builtin_amdgcn_wmma_f32_16x16x32_f16(
            false, a0.v, false, bf.v, (short)0, acc0[nt], false, false);
        acc1[nt] = __builtin_amdgcn_wmma_f32_16x16x32_f16(
            false, a1.v, false, bf.v, (short)0, acc1[nt], false, false);
      }
      __syncthreads();
    }
  }

  // ---- store to scratch in GEMM-natural (b, band, t, o) layout:
  // lanes 0-15 cover 16 consecutive o -> 64B-contiguous per half-wave store.
  const int rowg = mtile * MTILE + wave * 32 + hi * 8;
  const int bS = rowg >> 10, tS = rowg & (T_DIM - 1);
  float* srow = scratch + ((size_t)((bS * NBANDS + band) * T_DIM) + tS) * OUTCH;
#pragma unroll
  for (int nt = 0; nt < 8; ++nt) {
    const int o = nt * 16 + n;
#pragma unroll
    for (int r = 0; r < 8; ++r) {
      srow[(size_t)r * OUTCH + o]        = acc0[nt][r];
      srow[(size_t)(r + 16) * OUTCH + o] = acc1[nt][r];
    }
  }
}

// ---- Pass 3: (b,k,t,o) -> (b,o,t,k). One (b,t) slice per workgroup.
// Both global sides are 128B-per-wave contiguous; LDS padded to 129 floats
// per row so the transposed read is bank-conflict free.
__global__ __launch_bounds__(256) void transpose_ko(
    const float* __restrict__ scratch, float* __restrict__ out) {
  __shared__ float tile[NBANDS][OUTCH + 1];
  const int bt = blockIdx.x;            // 0..4095
  const int b = bt >> 10, t = bt & (T_DIM - 1);
  const int tid = threadIdx.x;
  {
    const int o = tid & (OUTCH - 1), k0 = tid >> 7;
    for (int k = k0; k < NBANDS; k += 2)
      tile[k][o] = scratch[((size_t)((b * NBANDS + k) * T_DIM) + t) * OUTCH + o];
  }
  __syncthreads();
  {
    const int k = tid & (NBANDS - 1), o0 = tid >> 6;
    for (int o = o0; o < OUTCH; o += 4)
      out[((size_t)((b * OUTCH + o) * T_DIM) + t) * NBANDS + k] = tile[k][o];
  }
}

extern "C" void kernel_launch(void* const* d_in, const int* in_sizes, int n_in,
                              void* d_out, int out_size, void* d_ws, size_t ws_size,
                              hipStream_t stream) {
  const float* x     = (const float*)d_in[0];
  const int*   idx   = (const int*)d_in[1];
  const float* mel_w = (const float*)d_in[2];
  const float* pre_w = (const float*)d_in[3];
  const float* pre_b = (const float*)d_in[4];
  float* out = (float*)d_out;

  const int W    = in_sizes[1] / NBANDS;      // taps per band (zero-padded)
  const int Kdim = 2 * W;
  const int Kpad = (Kdim + 31) & ~31;         // <= KPAD_MAX since W <= FBINS

  // workspace: [ f16 folded weights | f32 scratch (b,k,t,o) ]
  _Float16* wc = (_Float16*)d_ws;
  size_t wcBytes = ((size_t)NBANDS * OUTCH * Kpad * sizeof(_Float16) + 255) & ~(size_t)255;
  float* scratch = (float*)((char*)d_ws + wcBytes);

  const int total = NBANDS * OUTCH * Kpad;
  prep_weights<<<(total + 255) / 256, 256, 0, stream>>>(mel_w, pre_w, wc, W, Kpad, total);

  band_gemm<<<dim3(4 * T_DIM / MTILE, NBANDS), 256, 0, stream>>>(
      x, idx, wc, pre_b, scratch, W, Kpad);

  transpose_ko<<<4 * T_DIM, 256, 0, stream>>>(scratch, out);
}
